// GATEMB_12936441496232
// MI455X (gfx1250) — matrix operand order repaired and checked by reading, not compiled, not run on hardware
//
#include <hip/hip_runtime.h>
#include <hip/hip_bf16.h>
#include <stdint.h>

#define GN 20000
#define GE 320000
#define GH 4
#define GD 256
#define GHD 1024
#define NEG_SLOPE 0.2f

// padded LDS row stride (elements) for bank-conflict-free fragment reads
#define LDSTRIDE 40

typedef __bf16 bf16t;
typedef __attribute__((ext_vector_type(16))) bf16t v16bf;
typedef __attribute__((ext_vector_type(8)))  float v8f;

union Frag16 { v16bf v; unsigned u[8]; };

__device__ __forceinline__ unsigned short f2bf(float f) {
  unsigned u = __float_as_uint(f);
  unsigned r = 0x7FFFu + ((u >> 16) & 1u);   // round-to-nearest-even
  return (unsigned short)((u + r) >> 16);
}

// ---------------- fp32 -> bf16 bits (optional fused ReLU) ----------------
__global__ void k_f2bf(const float* __restrict__ in, unsigned short* __restrict__ out,
                       int n, int do_relu) {
  int i = blockIdx.x * blockDim.x + threadIdx.x;
  if (i < n) {
    float v = in[i];
    if (do_relu) v = fmaxf(v, 0.0f);
    out[i] = f2bf(v);
  }
}

// ---- fp32 W[K,Nn] -> bf16 Wt[Nn,K] (transpose once per layer) ----------
__global__ void k_wT(const float* __restrict__ W, unsigned short* __restrict__ Wt,
                     int K, int Nn) {
  int i = blockIdx.x * blockDim.x + threadIdx.x;
  if (i >= K * Nn) return;
  int n = i / K, k = i - n * K;
  Wt[i] = f2bf(W[(size_t)k * Nn + n]);
}

// ---------------- WMMA bf16 GEMM: C[M,Nn] = A[M,K] * Bt[Nn,K]^T ---------
// block: 256 threads = 8 waves (4 x 2); tile 128(M) x 128(N) x 32(K);
// each wave computes 32x64 = 8 x v_wmma_f32_16x16x32_bf16 per K-step.
__global__ __launch_bounds__(256) void k_gemm(const unsigned short* __restrict__ A,
                                              const unsigned short* __restrict__ Bt,
                                              float* __restrict__ C,
                                              int M, int K, int Nn) {
  __shared__ __align__(16) unsigned short As[128 * LDSTRIDE];  // [row][k], padded
  __shared__ __align__(16) unsigned short Bs[128 * LDSTRIDE];  // [n][k],  padded

  const int tid  = threadIdx.x;
  const int wave = tid >> 5;
  const int lane = tid & 31;
  const int wm   = wave & 3;   // m offset = 32*wm
  const int wn   = wave >> 2;  // n offset = 64*wn
  const int mBase = blockIdx.y * 128;
  const int nBase = blockIdx.x * 128;

  const int m16 = lane & 15;
  const int hi  = lane >> 4;

  v8f zero8 = {};
  v8f cacc[2][4];
#pragma unroll
  for (int am = 0; am < 2; ++am)
#pragma unroll
    for (int bn = 0; bn < 4; ++bn) cacc[am][bn] = zero8;

  const int r    = tid >> 1;          // staging row 0..127
  const int cseg = (tid & 1) * 16;    // staging k-segment {0,16}

  for (int k0 = 0; k0 < K; k0 += 32) {
    // stage A tile: 128x32 bf16, 16 elements (2 x uint4) per thread
    {
      int gm = mBase + r;
      uint4* dstp = reinterpret_cast<uint4*>(&As[r * LDSTRIDE + cseg]);
      if (gm < M) {
        const uint4* sp = reinterpret_cast<const uint4*>(A + (size_t)gm * K + k0 + cseg);
        dstp[0] = sp[0]; dstp[1] = sp[1];
      } else {
        uint4 z = {0u, 0u, 0u, 0u};
        dstp[0] = z; dstp[1] = z;
      }
    }
    // stage B tile (already transposed in global): rows nBase..nBase+127
    {
      const uint4* sp = reinterpret_cast<const uint4*>(Bt + (size_t)(nBase + r) * K + k0 + cseg);
      uint4* dstp = reinterpret_cast<uint4*>(&Bs[r * LDSTRIDE + cseg]);
      dstp[0] = sp[0]; dstp[1] = sp[1];
    }
    __syncthreads();

    // A fragments (16-bit A 16x32 layout): VGPR p holds K = (p>=4?16:0)+(p&3)*2+hi*8 (+1)
    Frag16 afr[2];
#pragma unroll
    for (int am = 0; am < 2; ++am) {
      int row = wm * 32 + am * 16 + m16;
#pragma unroll
      for (int p = 0; p < 8; ++p) {
        int kk = ((p >= 4) ? 16 : 0) + (p & 3) * 2 + hi * 8;
        afr[am].u[p] = *reinterpret_cast<const unsigned*>(&As[row * LDSTRIDE + kk]);
      }
    }
    // B fragments (16-bit B 32x16 layout): VGPR p holds K = (p>=4?16:0)+(p&3)*2+hi*8 (+1), N = lane&15
    Frag16 bfr[4];
#pragma unroll
    for (int bn = 0; bn < 4; ++bn) {
      int col = wn * 64 + bn * 16 + m16;
#pragma unroll
      for (int p = 0; p < 8; ++p) {
        int kk = ((p >= 4) ? 16 : 0) + (p & 3) * 2 + hi * 8;
        bfr[bn].u[p] = *reinterpret_cast<const unsigned*>(&Bs[col * LDSTRIDE + kk]);
      }
    }

#pragma unroll
    for (int am = 0; am < 2; ++am)
#pragma unroll
      for (int bn = 0; bn < 4; ++bn)
        cacc[am][bn] = __builtin_amdgcn_wmma_f32_16x16x32_bf16(
            false, afr[am].v, false, bfr[bn].v, (short)0, cacc[am][bn], false, false);

    __syncthreads();
  }

  // C layout: VGPR rr -> M = rr + 8*hi, N = lane&15
#pragma unroll
  for (int am = 0; am < 2; ++am)
#pragma unroll
    for (int bn = 0; bn < 4; ++bn)
#pragma unroll
      for (int rr = 0; rr < 8; ++rr) {
        int m  = wm * 32 + am * 16 + rr + 8 * hi;
        int n  = wn * 64 + bn * 16 + m16;
        int gm = mBase + m;
        if (gm < M) C[(size_t)gm * Nn + nBase + n] = cacc[am][bn][rr];
      }
}

// ---------------- per-node attention scores el/er -----------------------
__global__ __launch_bounds__(256) void k_scores(const float* __restrict__ fs,
                                                const float* __restrict__ fd,
                                                const float* __restrict__ al,
                                                const float* __restrict__ ar,
                                                float* __restrict__ el,
                                                float* __restrict__ er) {
  __shared__ float sd[256];
  int nId = blockIdx.x;
  int t = threadIdx.x;
  int hg = t >> 6, ln = t & 63;

  const float* fsrow = fs + (size_t)nId * GHD + hg * GD;
  const float* alr = al + hg * GD;
  float s = 0.0f;
  for (int j = ln; j < GD; j += 64) s += fsrow[j] * alr[j];
  sd[t] = s; __syncthreads();
  for (int off = 32; off > 0; off >>= 1) { if (ln < off) sd[t] += sd[t + off]; __syncthreads(); }
  if (ln == 0) el[nId * GH + hg] = sd[t];
  __syncthreads();

  const float* fdrow = fd + (size_t)nId * GHD + hg * GD;
  const float* arr = ar + hg * GD;
  s = 0.0f;
  for (int j = ln; j < GD; j += 64) s += fdrow[j] * arr[j];
  sd[t] = s; __syncthreads();
  for (int off = 32; off > 0; off >>= 1) { if (ln < off) sd[t] += sd[t + off]; __syncthreads(); }
  if (ln == 0) er[nId * GH + hg] = sd[t];
}

__global__ void k_init_mden(unsigned* __restrict__ mbits, float* __restrict__ den, int n) {
  int i = blockIdx.x * blockDim.x + threadIdx.x;
  if (i < n) { mbits[i] = 0u; den[i] = 0.0f; }
}

// pass 1: leaky-relu score per edge + segment max via ordered-uint atomicMax
__global__ void k_edge1(const float* __restrict__ el, const float* __restrict__ er,
                        const int* __restrict__ src, const int* __restrict__ dst,
                        float* __restrict__ ev, unsigned* __restrict__ mbits) {
  int i = blockIdx.x * blockDim.x + threadIdx.x;
  if (i >= GE * GH) return;
  int e = i >> 2, h = i & 3;
  float v = el[src[e] * GH + h] + er[dst[e] * GH + h];
  v = (v > 0.0f) ? v : NEG_SLOPE * v;
  ev[i] = v;
  unsigned u = __float_as_uint(v);
  u = (u & 0x80000000u) ? ~u : (u | 0x80000000u);
  atomicMax(&mbits[dst[e] * GH + h], u);
}

// pass 2: ex = exp(e - m[dst]); segment sum into den
__global__ void k_edge2(float* __restrict__ ev, const unsigned* __restrict__ mbits,
                        float* __restrict__ den, const int* __restrict__ dst) {
  int i = blockIdx.x * blockDim.x + threadIdx.x;
  if (i >= GE * GH) return;
  int e = i >> 2, h = i & 3;
  int d = dst[e];
  unsigned u = mbits[d * GH + h];
  float m = 0.0f;  // u==0 <=> empty segment <=> reference's isfinite fixup
  if (u) m = (u & 0x80000000u) ? __uint_as_float(u ^ 0x80000000u) : __uint_as_float(~u);
  float ex = expf(ev[i] - m);
  ev[i] = ex;
  atomicAdd(&den[d * GH + h], ex);
}

// pass 3: alpha = ex / den[dst] (in place)
__global__ void k_alpha(float* __restrict__ ev, const float* __restrict__ den,
                        const int* __restrict__ dst) {
  int i = blockIdx.x * blockDim.x + threadIdx.x;
  if (i >= GE * GH) return;
  int e = i >> 2, h = i & 3;
  ev[i] = ev[i] / den[dst[e] * GH + h];
}

__global__ void k_bias(float* __restrict__ acc, const float* __restrict__ b, int total) {
  int i = blockIdx.x * blockDim.x + threadIdx.x;
  if (i < total) acc[i] = b[i & (GHD - 1)];
}

// one block per edge: acc[dst] += fs[src] * alpha  (1024 floats)
__global__ __launch_bounds__(256) void k_agg(const float* __restrict__ fs,
                                             const float* __restrict__ alpha,
                                             const int* __restrict__ src,
                                             const int* __restrict__ dst,
                                             float* __restrict__ acc) {
  int e = blockIdx.x;
  int s = src[e], d = dst[e];
  const float* fsrow = fs + (size_t)s * GHD;
  float* arow = acc + (size_t)d * GHD;
  float av[GH];
#pragma unroll
  for (int j = 0; j < GH; ++j) av[j] = alpha[e * GH + j];
  int t = threadIdx.x;
#pragma unroll
  for (int j = 0; j < 4; ++j) {          // c = t + 256*j  =>  head == j
    int c = t + 256 * j;
    atomicAdd(&arow[c], fsrow[c] * av[j]);
  }
}

__global__ void k_mean(const float* __restrict__ acc, float* __restrict__ out) {
  int i = blockIdx.x * blockDim.x + threadIdx.x;
  if (i >= GN * GD) return;
  int n = i >> 8, d = i & 255;
  const float* r = acc + (size_t)n * GHD;
  out[i] = 0.25f * (r[d] + r[GD + d] + r[2 * GD + d] + r[3 * GD + d]);
}

// ------------------------------ launcher --------------------------------
extern "C" void kernel_launch(void* const* d_in, const int* in_sizes, int n_in,
                              void* d_out, int out_size, void* d_ws, size_t ws_size,
                              hipStream_t stream) {
  (void)in_sizes; (void)n_in; (void)out_size;

  const float* x   = (const float*)d_in[0];
  const int*   src = (const int*)d_in[1];
  const int*   dst = (const int*)d_in[2];
  const float* Wsrc[3] = {(const float*)d_in[3], (const float*)d_in[8],  (const float*)d_in[13]};
  const float* Wdst[3] = {(const float*)d_in[4], (const float*)d_in[9],  (const float*)d_in[14]};
  const float* al[3]   = {(const float*)d_in[5], (const float*)d_in[10], (const float*)d_in[15]};
  const float* ar[3]   = {(const float*)d_in[6], (const float*)d_in[11], (const float*)d_in[16]};
  const float* bb[3]   = {(const float*)d_in[7], (const float*)d_in[12], (const float*)d_in[17]};

  char* p = (char*)d_ws;
  auto alloc = [&](size_t bytes) -> void* {
    void* r = (void*)p;
    p += (bytes + 255) & ~(size_t)255;
    return r;
  };
  float* big[3];
  big[0] = (float*)alloc((size_t)GN * GHD * 4);
  big[1] = (float*)alloc((size_t)GN * GHD * 4);
  big[2] = (float*)alloc((size_t)GN * GHD * 4);
  unsigned short* hbf  = (unsigned short*)alloc((size_t)GN * GHD * 2);
  unsigned short* wsbf = (unsigned short*)alloc((size_t)1024 * GHD * 2);  // Wsrc^T bf16
  unsigned short* wdbf = (unsigned short*)alloc((size_t)1024 * GHD * 2);  // Wdst^T bf16
  float*    elb = (float*)alloc((size_t)GN * GH * 4);
  float*    erb = (float*)alloc((size_t)GN * GH * 4);
  unsigned* mbb = (unsigned*)alloc((size_t)GN * GH * 4);
  float*    den = (float*)alloc((size_t)GN * GH * 4);
  float*    ev  = (float*)alloc((size_t)GE * GH * 4);
  if ((size_t)(p - (char*)d_ws) > ws_size) return;  // ws too small: bail

  const int fins[3]   = {512, 1024, 1024};
  const int fsIdx[3]  = {0, 0, 0};
  const int accIdx[3] = {1, 2, 1};

  const float* hin = x;
  for (int l = 0; l < 3; ++l) {
    const int Fin = fins[l];
    float* fsb  = big[fsIdx[l]];
    float* accb = big[accIdx[l]];

    int nh = GN * Fin;
    k_f2bf<<<(nh + 255) / 256, 256, 0, stream>>>(hin, hbf, nh, l > 0 ? 1 : 0);
    int nw = Fin * GHD;
    k_wT<<<(nw + 255) / 256, 256, 0, stream>>>(Wsrc[l], wsbf, Fin, GHD);
    k_wT<<<(nw + 255) / 256, 256, 0, stream>>>(Wdst[l], wdbf, Fin, GHD);

    dim3 gg(GHD / 128, (GN + 127) / 128);
    k_gemm<<<gg, 256, 0, stream>>>(hbf, wsbf, fsb,  GN, Fin, GHD);
    k_gemm<<<gg, 256, 0, stream>>>(hbf, wdbf, accb, GN, Fin, GHD);

    k_scores<<<GN, 256, 0, stream>>>(fsb, accb, al[l], ar[l], elb, erb);

    int nh2 = GN * GH;
    k_init_mden<<<(nh2 + 255) / 256, 256, 0, stream>>>(mbb, den, nh2);
    int ne = GE * GH;
    k_edge1<<<(ne + 255) / 256, 256, 0, stream>>>(elb, erb, src, dst, ev, mbb);
    k_edge2<<<(ne + 255) / 256, 256, 0, stream>>>(ev, mbb, den, dst);
    k_alpha<<<(ne + 255) / 256, 256, 0, stream>>>(ev, den, dst);

    int tot = GN * GHD;
    k_bias<<<(tot + 255) / 256, 256, 0, stream>>>(accb, bb[l], tot);
    k_agg<<<GE, 256, 0, stream>>>(fsb, ev, src, dst, accb);

    hin = accb;
  }

  k_mean<<<(GN * GD + 255) / 256, 256, 0, stream>>>(big[1], (float*)d_out);
}
